// HadamardTrustQuantizer_52424370815007
// MI455X (gfx1250) — matrix operand order repaired and checked by reading, not compiled, not run on hardware
//
#include <hip/hip_runtime.h>

// MI455X / gfx1250: wave32, WMMA 16x16x32 f16 -> f32.
// out = HadRot( quantize( HadRot(x) ) ), Hadamard done as f16 WMMA with +-1
// entries (exact); 1/sqrt(128) folded into per-row scalars. GEMM2 operates on
// exact integer levels in f16, so it is exact.

typedef __attribute__((ext_vector_type(16))) _Float16 v16h;
typedef __attribute__((ext_vector_type(4)))  _Float16 v4h;
typedef __attribute__((ext_vector_type(8)))  float    v8f;
typedef __attribute__((ext_vector_type(4)))  float    v4f;

#define COLS     4096
#define STR      36                      // LDS row stride in halfs (72B): 8B-aligned, conflict-free frag reads
#define R128INV  0.08838834764831845f    // 1/sqrt(128)

union Frag16 { v16h h; unsigned int u[8]; };

__device__ __forceinline__ v8f wmma_f16(v16h a, v16h b, v8f c) {
  // (neg_a, A, neg_b, B, c_mod, C, reuse_a, reuse_b)
  return __builtin_amdgcn_wmma_f32_16x16x32_f16(false, a, false, b, (short)0, c, false, false);
}

// A-fragment (16x32 f16) from LDS tile (row stride STR halfs), contiguous k order (k = 16h + s):
// lanes 0-15: M=lane, K {0..15}; lanes 16-31: M=lane-16, K {16..31}
__device__ __forceinline__ v16h load_a_frag(const _Float16* st, int m, int khOff) {
  Frag16 A;
#pragma unroll
  for (int i = 0; i < 4; ++i) {
    A.u[i]     = *(const unsigned int*)(st + m * STR + khOff + 2 * i);
    A.u[i + 4] = *(const unsigned int*)(st + m * STR + 8 + khOff + 2 * i);
  }
  return A.h;
}

// B fragment = per-lane base fragment XOR a whole-fragment sign mask.
__device__ __forceinline__ v16h make_b_frag(const unsigned int* Bn, unsigned int msk) {
  Frag16 B;
#pragma unroll
  for (int i = 0; i < 8; ++i) B.u[i] = Bn[i] ^ msk;
  return B.h;
}

__global__ __launch_bounds__(1024, 1)
void hadq_fwd_kernel(const float* __restrict__ x, float* __restrict__ out) {
  __shared__ __align__(16) _Float16 stageAll[32 * 16 * STR];  // ~36 KB: per-wave 16x32 f16 restage tile
  __shared__ float rowSumSq[16];
  __shared__ float invStepS[16];
  __shared__ float outScaleS[16];

  const int tid   = threadIdx.x;
  const int w     = tid >> 5;        // wave id: owns 128-col Hadamard block w
  const int lane  = tid & 31;
  const int n     = lane & 15;       // B/C/D column within 16-tile
  const int lhi   = lane >> 4;       // lane half
  const int m     = n;               // A row for this lane
  const int khOff = lhi * 16;

  _Float16* st = stageAll + w * (16 * STR);

  const long long gbase = (long long)blockIdx.x * 16 * COLS;  // 16 rows per WG
  const int cb = w * 128;                                     // this wave's column block

  if (tid < 16) rowSumSq[tid] = 0.0f;
  __syncthreads();

  // Base B fragment: H[k][ncol] sign = parity(popc(k & ncol)); low 4 bits of
  // ncol are n, and the element index i2 (0..15) only has low-4 bits, so the
  // base depends only on n. K-offset bits (q<<5) and lane-half bit (lhi<<4)
  // contribute a single whole-fragment sign flip per (j, q).
  unsigned int Bn[8];
#pragma unroll
  for (int i = 0; i < 8; ++i) {
    unsigned lo = (unsigned)(__popc((2 * i)     & n) & 1);
    unsigned hi = (unsigned)(__popc((2 * i + 1) & n) & 1);
    Bn[i] = 0x3C003C00u | (lo << 15) | (hi << 31);   // +-1.0 in f16, packed
  }

  // ---------------- GEMM1: c = X_blk(16x128) @ Hs(128x128), Hs = +-1 -------
  v8f c[8];
#pragma unroll
  for (int j = 0; j < 8; ++j)
#pragma unroll
    for (int e = 0; e < 8; ++e) c[j][e] = 0.0f;

  for (int q = 0; q < 4; ++q) {       // K chunks of 32
    const float* gsrc = x + gbase + cb + q * 32;
#pragma unroll
    for (int r = 0; r < 4; ++r) {     // coalesced 16x32 f32 -> f16 LDS stage
      int f   = r * 32 + lane;
      int row = f >> 3;
      int c4  = f & 7;
      v4f v = *(const v4f*)(gsrc + (long long)row * COLS + c4 * 4);
      v4h hv;
#pragma unroll
      for (int e = 0; e < 4; ++e) hv[e] = (_Float16)v[e];
      *(v4h*)(st + row * STR + c4 * 4) = hv;
    }
    v16h a = load_a_frag(st, m, khOff);
#pragma unroll
    for (int j = 0; j < 8; ++j) {
      unsigned flip = ((unsigned)(lhi & (j & 1))) ^ (unsigned)(__popc(q & (j >> 1)) & 1);
      v16h bf = make_b_frag(Bn, flip * 0x80008000u);
      c[j] = wmma_f16(a, bf, c[j]);
    }
  }

  // ---------------- per-row sum of squares (c holds x_h * sqrt(128)) -------
  float sq[8];
#pragma unroll
  for (int r = 0; r < 8; ++r) sq[r] = 0.0f;
#pragma unroll
  for (int j = 0; j < 8; ++j)
#pragma unroll
    for (int r = 0; r < 8; ++r) sq[r] += c[j][r] * c[j][r];
#pragma unroll
  for (int xm = 1; xm <= 8; xm <<= 1)
#pragma unroll
    for (int r = 0; r < 8; ++r) sq[r] += __shfl_xor(sq[r], xm, 32);
  if (n == 0) {
#pragma unroll
    for (int r = 0; r < 8; ++r) atomicAdd(&rowSumSq[r + 8 * lhi], sq[r]);
  }
  __syncthreads();

  if (tid < 16) {
    // std = max(sqrt(mean(x_h^2)), 1e-8); x_h^2 = c^2/128, mean over 4096
    float ms   = rowSumSq[tid] * (1.0f / 524288.0f);
    float sd   = fmaxf(sqrtf(ms), 1e-8f);
    float step = (2.5139f / 7.0f) * sd;
    invStepS[tid]  = R128INV / step;   // q = rint(c * invStep)
    outScaleS[tid] = step * R128INV;   // out = (q @ Hs) * step / sqrt(128)
  }
  __syncthreads();

  float invStep[8], oSc[8];
#pragma unroll
  for (int r = 0; r < 8; ++r) {
    invStep[r] = invStepS[r + 8 * lhi];
    oSc[r]     = outScaleS[r + 8 * lhi];
  }

  // ---------------- quantize + GEMM2: d = Q(16x128) @ Hs(128x128) ----------
  v8f d[8];
#pragma unroll
  for (int j = 0; j < 8; ++j)
#pragma unroll
    for (int e = 0; e < 8; ++e) d[j][e] = 0.0f;

  for (int hq = 0; hq < 4; ++hq) {    // K chunks of 32 (2 output tiles of GEMM1)
#pragma unroll
    for (int t = 0; t < 2; ++t) {     // C-layout -> A-layout via LDS restage
      int jj = hq * 2 + t;
#pragma unroll
      for (int r = 0; r < 8; ++r) {
        float qv = rintf(c[jj][r] * invStep[r]);
        qv = fminf(7.0f, fmaxf(-7.0f, qv));          // exact integer level in f16
        st[(r + 8 * lhi) * STR + t * 16 + n] = (_Float16)qv;
      }
    }
    v16h a = load_a_frag(st, m, khOff);
#pragma unroll
    for (int j = 0; j < 8; ++j) {
      unsigned flip = ((unsigned)(lhi & (j & 1))) ^ (unsigned)(__popc(hq & (j >> 1)) & 1);
      v16h bf = make_b_frag(Bn, flip * 0x80008000u);
      d[j] = wmma_f16(a, bf, d[j]);
    }
  }

  // ---------------- scale per row and store (coalesced 64B per lane-half) --
#pragma unroll
  for (int j = 0; j < 8; ++j)
#pragma unroll
    for (int r = 0; r < 8; ++r)
      out[gbase + (long long)(r + 8 * lhi) * COLS + cb + j * 16 + n] = d[j][r] * oSc[r];
}

extern "C" void kernel_launch(void* const* d_in, const int* in_sizes, int n_in,
                              void* d_out, int out_size, void* d_ws, size_t ws_size,
                              hipStream_t stream) {
  (void)n_in; (void)out_size; (void)d_ws; (void)ws_size;
  const float* x = (const float*)d_in[0];
  (void)d_in[1];  // H unused: Sylvester Hadamard signs are generated in-kernel
  float* out = (float*)d_out;

  long long total = (long long)in_sizes[0];      // 4*4096*4096
  int rows = (int)(total / COLS);                // 16384
  dim3 grid(rows / 16);                          // 16 rows per workgroup
  dim3 block(1024);                              // 32 waves, 1 Hadamard block each
  hipLaunchKernelGGL(hadq_fwd_kernel, grid, block, 0, stream, x, out);
}
